// ScaledDotProductAttention_Pytorch_50483045597399
// MI455X (gfx1250) — compile-verified, matrix operand order checked by reading, not run on hardware
//
#include <hip/hip_runtime.h>

// ---------------------------------------------------------------------------
// Fused attention (context + full normalized attn output) for gfx1250.
//   scores = exp(Q K^T / sqrt(64)) * mask ; attn = scores/(rowsum+1e-8)
//   context = attn @ V
// One workgroup (4 x wave32) handles a 16-query tile of one (b,h) pair and
// keeps the full 16x2048 exp-score strip resident in LDS (128 KB) so the
// 1 GiB attn tensor is written to HBM exactly once (non-temporal).
// Phase 3 streams V through LDS with CDNA5 async global->LDS copies
// (ASYNCcnt / s_wait_asynccnt), double-buffered against the WMMAs.
// ---------------------------------------------------------------------------

typedef __attribute__((ext_vector_type(16))) _Float16 v16h;
typedef __attribute__((ext_vector_type(8)))  float    v8f;
typedef int vint4 __attribute__((vector_size(16)));   // builtin's b128 unit

#define S_LEN  2048
#define D_DIM  64
#define BH     64          // B*H = 4*16
#define M_TILE 16
#define NTHREADS 128
#define SCALE  0.125f      // 1/sqrt(64)
#define LOG2E  1.44269504088896340736f

#if defined(__has_builtin)
#  if __has_builtin(__builtin_amdgcn_global_load_async_to_lds_b128)
#    define HAS_ASYNC_LDS 1
#  else
#    define HAS_ASYNC_LDS 0
#  endif
#else
#  define HAS_ASYNC_LDS 0
#endif

// 16-byte global -> LDS copy. Async (no VGPR round-trip, ASYNCcnt) when the
// toolchain exposes the gfx1250 builtin; synchronous vector copy otherwise.
__device__ inline void copy16_g2l(const float* gsrc, float* ldst) {
#if HAS_ASYNC_LDS
    __builtin_amdgcn_global_load_async_to_lds_b128(
        (__attribute__((address_space(1))) vint4*)const_cast<float*>(gsrc),
        (__attribute__((address_space(3))) vint4*)ldst,
        /*offset=*/0, /*cpol=*/0);
#else
    *(float4*)ldst = *(const float4*)gsrc;
#endif
}

__device__ inline void async_wait0() {
#if HAS_ASYNC_LDS
#  if __has_builtin(__builtin_amdgcn_s_wait_asynccnt)
    __builtin_amdgcn_s_wait_asynccnt(0);
#  else
    asm volatile("s_wait_asynccnt 0x0" ::: "memory");
#  endif
#endif
}

// ---- fragment loaders (CDNA5 16-bit WMMA layouts, ISA 7.12.2) -------------
// A 16x32 f16: lane&15 = row M; lanes 0-15 hold k {0..7,16..23}, lanes 16-31
// hold k {8..15,24..31}; halves h<8 -> k=bk+h, h>=8 -> k=16+bk+(h-8).
__device__ inline v16h load_a_lds_f16(const _Float16* src, int kbase, int lane) {
    const int m  = lane & 15;
    const int bk = (lane < 16) ? 0 : 8;
    const _Float16* r = src + m * D_DIM + kbase + bk;
    v16h a;
#pragma unroll
    for (int h = 0; h < 8; ++h) { a[h] = r[h]; a[h + 8] = r[16 + h]; }
    return a;
}

// A fragment built from the f32 exp-score strip in LDS (convert on the fly).
__device__ inline v16h load_a_lds_f32(const float* sp, int kk, int lane) {
    const int m  = lane & 15;
    const int bk = (lane < 16) ? 0 : 8;
    const float* r = sp + m * S_LEN + kk + bk;
    v16h a;
#pragma unroll
    for (int h = 0; h < 8; ++h) {
        a[h]     = (_Float16)r[h];
        a[h + 8] = (_Float16)r[16 + h];
    }
    return a;
}

// B 32x16 f16 for Q*K^T: B[k][n] = K[key n][k]; lane&15 = column n.
// Each lane reads two contiguous 8-float runs of one K row (L2-friendly).
__device__ inline v16h load_b_kT(const float* Kb, int kr0, int kbase, int lane) {
    const int n  = lane & 15;
    const int bk = (lane < 16) ? 0 : 8;
    const float* r = Kb + (size_t)(kr0 + n) * D_DIM + kbase + bk;
    v16h b;
#pragma unroll
    for (int h = 0; h < 8; ++h) {
        b[h]     = (_Float16)r[h];
        b[h + 8] = (_Float16)r[16 + h];
    }
    return b;
}

// B 32x16 f16 for P*V from the f32 V slab in LDS: B[k][n] = sV[k][n0+n].
__device__ inline v16h load_b_v32(const float* sv, int n0, int lane) {
    const int n  = n0 + (lane & 15);
    const int bk = (lane < 16) ? 0 : 8;
    v16h b;
#pragma unroll
    for (int h = 0; h < 8; ++h) {
        b[h]     = (_Float16)sv[(bk + h) * D_DIM + n];
        b[h + 8] = (_Float16)sv[(16 + bk + h) * D_DIM + n];
    }
    return b;
}

__global__ __launch_bounds__(NTHREADS)
void attn_fused_kernel(const float* __restrict__ Q, const float* __restrict__ K,
                       const float* __restrict__ V, const float* __restrict__ mask,
                       float* __restrict__ ctx, float* __restrict__ attn)
{
    __shared__ float    sP[M_TILE * S_LEN];   // 128 KB: exp-score strip
    __shared__ _Float16 sQ[M_TILE * D_DIM];   // 2 KB: pre-scaled Q tile (f16)
    __shared__ float    sV[2][32 * D_DIM];    // 2 x 8 KB: V k-slabs (f32, DMA)
    __shared__ float    sRow[M_TILE];
    __shared__ float    sInv[M_TILE];

    const int tid  = threadIdx.x;
    const int lane = tid & 31;
    const int wave = tid >> 5;
    const int q0   = blockIdx.x * M_TILE;
    const int bh   = blockIdx.y;

    const float* Qb = Q + (size_t)bh * S_LEN * D_DIM;
    const float* Kb = K + (size_t)bh * S_LEN * D_DIM;
    const float* Vb = V + (size_t)bh * S_LEN * D_DIM;

    if (tid < M_TILE) sRow[tid] = 0.0f;

    // Stage Q tile into LDS as f16, folding in the 1/sqrt(d) scale.
#pragma unroll
    for (int i = 0; i < (M_TILE * D_DIM) / NTHREADS; ++i) {
        int idx = tid + i * NTHREADS;
        sQ[idx] = (_Float16)(Qb[(size_t)(q0 + (idx >> 6)) * D_DIM + (idx & 63)] * SCALE);
    }
    __syncthreads();

    const v16h aQ0 = load_a_lds_f16(sQ, 0,  lane);
    const v16h aQ1 = load_a_lds_f16(sQ, 32, lane);

    // -------- Phase 1: S = exp(Q K^T / sqrt(d)) into LDS -------------------
    // 4 waves split the 128 key tiles (uniform 32 trips each; EXEC all-ones).
    for (int j = wave; j < S_LEN / 16; j += 4) {
        const int kr0 = j * 16;
        const v16h b0 = load_b_kT(Kb, kr0, 0,  lane);
        const v16h b1 = load_b_kT(Kb, kr0, 32, lane);
        v8f acc = {};
        acc = __builtin_amdgcn_wmma_f32_16x16x32_f16(false, aQ0, false, b0,
                                                     (short)0, acc, false, false);
        acc = __builtin_amdgcn_wmma_f32_16x16x32_f16(false, aQ1, false, b1,
                                                     (short)0, acc, false, false);
        const int mb = (lane < 16) ? 0 : 8;   // C layout: VGPR r -> M = r + mb
        const int n  = lane & 15;
#pragma unroll
        for (int r = 0; r < 8; ++r) {
            float e = __builtin_amdgcn_exp2f(acc[r] * LOG2E);  // v_exp_f32
            sP[(mb + r) * S_LEN + kr0 + n] = e;
        }
    }
    __syncthreads();

    // -------- Phase 2a: apply mask + row sums ------------------------------
    {
        const int row = tid >> 3;            // 8 threads per query row
        const int seg = tid & 7;             // each sums 256 columns
        const float* mrow = mask + (size_t)(q0 + row) * S_LEN;
        float* prow = sP + row * S_LEN;
        float local = 0.0f;
        const int c0 = seg * (S_LEN / 8);
        for (int c = c0; c < c0 + (S_LEN / 8); ++c) {
            float p = prow[c] * mrow[c];
            prow[c] = p;
            local  += p;
        }
        atomicAdd(&sRow[row], local);
    }
    __syncthreads();
    if (tid < M_TILE) sInv[tid] = 1.0f / (sRow[tid] + 1e-8f);
    __syncthreads();

    // -------- Phase 2b: normalize in LDS, stream attn (non-temporal) -------
    float* attnRow = attn + ((size_t)bh * S_LEN + q0) * S_LEN;
    for (int i = 0; i < (M_TILE * S_LEN) / NTHREADS; ++i) {
        int idx = tid + i * NTHREADS;        // row = idx>>11, col = idx&2047
        float p = sP[idx] * sInv[idx >> 11];
        sP[idx] = p;
        __builtin_nontemporal_store(p, &attnRow[idx]);  // TH=NT stream-out
    }
    __syncthreads();

    // -------- Phase 3: context = P @ V, async double-buffered V slabs ------
    // One 32x64 f32 slab is 2048 contiguous floats: 128 threads x 4 x b128.
    const int n0 = wave * 16;                // each wave owns 16 of the 64 dims
    v8f oacc = {};

#pragma unroll
    for (int c = 0; c < 4; ++c) {            // prologue: DMA slab 0
        int e4 = (tid + c * NTHREADS) * 4;
        copy16_g2l(Vb + e4, &sV[0][e4]);
    }

    for (int i = 0; i < S_LEN / 32; ++i) {
        const int cur = i & 1;
        async_wait0();                       // our slab-i copies complete
        __syncthreads();                     // visible to all; buf cur^1 free
        if (i + 1 < S_LEN / 32) {            // DMA next slab under the WMMA
#pragma unroll
            for (int c = 0; c < 4; ++c) {
                int e4 = (tid + c * NTHREADS) * 4;
                copy16_g2l(Vb + (size_t)(i + 1) * 32 * D_DIM + e4,
                           &sV[cur ^ 1][e4]);
            }
        }
        const v16h aP = load_a_lds_f32(sP, i * 32, lane);
        const v16h bV = load_b_v32(sV[cur], n0, lane);
        oacc = __builtin_amdgcn_wmma_f32_16x16x32_f16(false, aP, false, bV,
                                                      (short)0, oacc, false, false);
    }

    {
        const int mb = (lane < 16) ? 0 : 8;
        const int n  = lane & 15;
        float* crow = ctx + ((size_t)bh * S_LEN + q0) * D_DIM;
#pragma unroll
        for (int r = 0; r < 8; ++r)
            crow[(size_t)(mb + r) * D_DIM + n0 + n] = oacc[r];
    }
}

extern "C" void kernel_launch(void* const* d_in, const int* in_sizes, int n_in,
                              void* d_out, int out_size, void* d_ws, size_t ws_size,
                              hipStream_t stream) {
    const float* Q    = (const float*)d_in[0];
    const float* K    = (const float*)d_in[1];
    const float* V    = (const float*)d_in[2];
    const float* mask = (const float*)d_in[3];

    float* ctx  = (float*)d_out;                                   // [B,H,S,64]
    float* attn = (float*)d_out + (size_t)BH * S_LEN * D_DIM;      // [B,H,S,S]

    dim3 grid(S_LEN / M_TILE, BH);
    attn_fused_kernel<<<grid, NTHREADS, 0, stream>>>(Q, K, V, mask, ctx, attn);
}
